// MultiheadAttention_83245056131965
// MI455X (gfx1250) — matrix operand order repaired and checked by reading, not compiled, not run on hardware
//
#include <hip/hip_runtime.h>

// ---------------------------------------------------------------------------
// Types for CDNA5 WMMA (wave32): D(16x16 f32) = A(16x32 bf16) x B(32x16 bf16) + C
// ---------------------------------------------------------------------------
typedef __bf16 bf16;
typedef __bf16 bf16x8  __attribute__((ext_vector_type(8)));
typedef __bf16 bf16x16 __attribute__((ext_vector_type(16)));
typedef float  f32x8   __attribute__((ext_vector_type(8)));

union AFrag { bf16x16 v; bf16x8 h[2]; };

__device__ __forceinline__ f32x8 wmma_bf16(bf16x16 a, bf16x16 b, f32x8 c) {
  // (neg_a, A, neg_b, B, c_mod, C, reuse_a, reuse_b)
  return __builtin_amdgcn_wmma_f32_16x16x32_bf16(false, a, false, b, (short)0, c,
                                                 false, false);
}

// CDNA5 async global->LDS copy (ASYNCcnt-tracked). lds = 32-bit LDS byte addr
// (low 32 bits of a generic pointer to __shared__), g = 64-bit global address.
__device__ __forceinline__ void async_b128(unsigned lds, const void* g) {
  unsigned long long ga = (unsigned long long)g;
  asm volatile("global_load_async_to_lds_b128 %0, %1, off"
               :: "v"(lds), "v"(ga)
               : "memory");
}
__device__ __forceinline__ void wait_async0() {
  asm volatile("s_wait_asynccnt 0x0" ::: "memory");
}

// Problem constants (fixed by the reference)
constexpr int Bsz = 4;
constexpr int T   = 2048;
constexpr int F   = 256;
constexpr int H   = 8;
constexpr int HF  = H * F;        // 2048
constexpr int HF3 = H * F * 3;    // 6144
constexpr int BT  = Bsz * T;      // 8192

// LDS padding strides (elements) to avoid bank conflicts on ds_load_b128
constexpr int KPAD = 264;   // 32 rows x 264 bf16 (528 B stride -> bank step 4)
constexpr int VPAD = 40;    // 256 rows x 40 bf16 (80 B stride -> bank step 20)
constexpr int PPAD = 40;    // 16 rows x 40 bf16 per wave

// Workspace layout (bytes)
constexpr size_t OFF_XB  = 0;                                   // x bf16 [8192,256]
constexpr size_t OFF_WQT = OFF_XB  + (size_t)BT * F * 2;        // Wqkv^T bf16 [6144,256]
constexpr size_t OFF_WPT = OFF_WQT + (size_t)HF3 * F * 2;       // Wproj^T bf16 [256,2048]
constexpr size_t OFF_Q   = OFF_WPT + (size_t)F * HF * 2;        // Q bf16 [B,H,T,F]
constexpr size_t OFF_K   = OFF_Q   + (size_t)Bsz * H * T * F * 2;
constexpr size_t OFF_VT  = OFF_K   + (size_t)Bsz * H * T * F * 2;   // V^T bf16 [B,H,F,T]
constexpr size_t OFF_AB  = OFF_VT  + (size_t)Bsz * H * T * F * 2;   // attn out bf16 [B,T,HF]

// ---------------------------------------------------------------------------
// Prep kernels (tiny; weights are L2-resident afterwards)
// ---------------------------------------------------------------------------
__global__ void k_cvt_f32_bf16(const float* __restrict__ in, bf16* __restrict__ out, int n) {
  int i = blockIdx.x * blockDim.x + threadIdx.x;
  if (i < n) out[i] = (bf16)in[i];
}

__global__ void k_tr_wqkv(const float* __restrict__ w, bf16* __restrict__ wt) {
  int c = blockIdx.x;
  int k = threadIdx.x;
  wt[c * F + k] = (bf16)w[(size_t)k * HF3 + c];
}

__global__ void k_tr_wproj(const float* __restrict__ w, bf16* __restrict__ wt) {
  int n = blockIdx.x;
  for (int k = threadIdx.x; k < HF; k += blockDim.x)
    wt[(size_t)n * HF + k] = (bf16)w[(size_t)k * F + n];
}

// ---------------------------------------------------------------------------
// Kernel A: QKV GEMM, scatter into Q/K (x 1/16) and V^T
// ---------------------------------------------------------------------------
__global__ void __launch_bounds__(128) k_qkv_gemm(const bf16* __restrict__ xb,
                                                  const bf16* __restrict__ Wqt,
                                                  const float* __restrict__ bqkv,
                                                  bf16* __restrict__ Qb,
                                                  bf16* __restrict__ Kb,
                                                  bf16* __restrict__ Vt) {
  const int wave = threadIdx.x >> 5;
  const int lane = threadIdx.x & 31;
  const int ln = lane & 15, lh = lane >> 4;
  const int m0 = (blockIdx.x * 4 + wave) * 16;
  const int c0 = blockIdx.y * 64;

  f32x8 acc[4];
#pragma unroll
  for (int t = 0; t < 4; ++t)
#pragma unroll
    for (int i = 0; i < 8; ++i) acc[t][i] = 0.0f;

#pragma unroll
  for (int ks = 0; ks < 8; ++ks) {
    const int k0 = ks * 32;
    AFrag a;
    const bf16* ap = xb + (size_t)(m0 + ln) * F + k0 + lh * 16;
    a.h[0] = *(const bf16x8*)ap;
    a.h[1] = *(const bf16x8*)(ap + 8);
#pragma unroll
    for (int t = 0; t < 4; ++t) {
      bf16x16 b = *(const bf16x16*)(Wqt + (size_t)(c0 + t * 16 + ln) * F + k0 + lh * 16);
      acc[t] = wmma_bf16(a.v, b, acc[t]);
    }
  }

#pragma unroll
  for (int t = 0; t < 4; ++t) {
    const int c = c0 + t * 16 + ln;
    const float bias = bqkv[c];
    const int which = c % 3;
    const int f = (c / 3) & (F - 1);
    const int h = c / (3 * F);
#pragma unroll
    for (int i = 0; i < 8; ++i) {
      const int m = m0 + i + 8 * lh;
      const int bb = m >> 11;
      const int tq = m & (T - 1);
      const float v = acc[t][i] + bias;
      if (which == 0) {
        Qb[((size_t)(bb * H + h) * T + tq) * F + f] = (bf16)(v * 0.0625f);
      } else if (which == 1) {
        Kb[((size_t)(bb * H + h) * T + tq) * F + f] = (bf16)(v * 0.0625f);
      } else {
        Vt[((size_t)(bb * H + h) * F + f) * T + tq] = (bf16)v;
      }
    }
  }
}

// ---------------------------------------------------------------------------
// Kernel B: flash attention. Block = 4 waves, all on the SAME (b,h); wave w
// handles 16 queries. K/V key-blocks of 32 are staged cooperatively into LDS
// with async global->LDS copies, double-buffered (copy overlaps WMMA).
// ---------------------------------------------------------------------------
__global__ void __launch_bounds__(128) k_attn(const bf16* __restrict__ Qb,
                                              const bf16* __restrict__ Kb,
                                              const bf16* __restrict__ Vt,
                                              bf16* __restrict__ Ab) {
  __shared__ __align__(16) bf16 sK[2][32 * KPAD];    // K block [32 keys][256 f] padded
  __shared__ __align__(16) bf16 sV[2][256 * VPAD];   // V^T block [256 f][32 keys] padded
  __shared__ __align__(16) bf16 pbuf[4 * 16 * PPAD]; // per-wave P tile 16x32 padded

  const int tid  = threadIdx.x;
  const int wave = tid >> 5;
  const int lane = tid & 31;
  const int ln = lane & 15, lh = lane >> 4;
  const int gw = blockIdx.x * 4 + wave;
  const int bh = gw >> 7;        // 128 q-tiles per (b,h); 4 waves share bh
  const int qt = gw & 127;

  const bf16* Qp = Qb + (size_t)bh * T * F + (size_t)qt * 16 * F;
  const bf16* Kp = Kb + (size_t)bh * T * F;
  const bf16* Vp = Vt + (size_t)bh * F * T;

  // Cooperative async stage of one 32-key block of K and V^T into LDS.
  auto stage = [&](int st, int s0) {
#pragma unroll
    for (int i = 0; i < 8; ++i) {           // K: 32 rows x 32 chunks of 16B
      const int c = tid + i * 128;          // c in [0,1024)
      const int r = c >> 5, cc = c & 31;
      async_b128((unsigned)(uintptr_t)&sK[st][r * KPAD + cc * 8],
                 Kp + (size_t)(s0 + r) * F + cc * 8);
    }
#pragma unroll
    for (int i = 0; i < 8; ++i) {           // V^T: 256 rows x 4 chunks of 16B
      const int c = tid + i * 128;
      const int f = c >> 2, p = c & 3;
      async_b128((unsigned)(uintptr_t)&sV[st][f * VPAD + p * 8],
                 Vp + (size_t)f * T + s0 + p * 8);
    }
  };

  // Resident Q fragments: 8 k-steps of 32
  AFrag qf[8];
#pragma unroll
  for (int ks = 0; ks < 8; ++ks) {
    const bf16* p = Qp + (size_t)ln * F + ks * 32 + lh * 16;
    qf[ks].h[0] = *(const bf16x8*)p;
    qf[ks].h[1] = *(const bf16x8*)(p + 8);
  }

  f32x8 oacc[16];
#pragma unroll
  for (int nt = 0; nt < 16; ++nt)
#pragma unroll
    for (int i = 0; i < 8; ++i) oacc[nt][i] = 0.0f;

  float mrow[8], lrow[8];
#pragma unroll
  for (int i = 0; i < 8; ++i) { mrow[i] = -__builtin_inff(); lrow[i] = 0.0f; }

  bf16* pb = pbuf + wave * (16 * PPAD);

  stage(0, 0);

  for (int j = 0; j < T / 32; ++j) {
    const int st = j & 1;
    const int s0 = j * 32;

    wait_async0();        // own wave's loads for buffer st are done
    __syncthreads();      // everyone's loads done; everyone done reading st^1
    if (j + 1 < T / 32) stage(st ^ 1, s0 + 32);  // prefetch next block

    // ---- S = Q @ K^T : two 16x16 f32 fragments ----
    f32x8 sa, sb;
#pragma unroll
    for (int i = 0; i < 8; ++i) { sa[i] = 0.0f; sb[i] = 0.0f; }
#pragma unroll
    for (int ks = 0; ks < 8; ++ks) {
      bf16x16 kf0 = *(const bf16x16*)&sK[st][ln * KPAD + ks * 32 + lh * 16];
      bf16x16 kf1 = *(const bf16x16*)&sK[st][(16 + ln) * KPAD + ks * 32 + lh * 16];
      sa = wmma_bf16(qf[ks].v, kf0, sa);
      sb = wmma_bf16(qf[ks].v, kf1, sb);
    }

    // ---- online softmax (rows live across 16-lane halves) ----
    float alpha[8];
#pragma unroll
    for (int i = 0; i < 8; ++i) {
      float mx = fmaxf(sa[i], sb[i]);
#pragma unroll
      for (int msk = 1; msk < 16; msk <<= 1) mx = fmaxf(mx, __shfl_xor(mx, msk, 32));
      const float mn = fmaxf(mrow[i], mx);
      const float al = __expf(mrow[i] - mn);
      const float pa = __expf(sa[i] - mn);
      const float pbv = __expf(sb[i] - mn);
      sa[i] = pa; sb[i] = pbv;
      float rs = pa + pbv;
#pragma unroll
      for (int msk = 1; msk < 16; msk <<= 1) rs += __shfl_xor(rs, msk, 32);
      lrow[i] = lrow[i] * al + rs;
      mrow[i] = mn;
      alpha[i] = al;
    }
#pragma unroll
    for (int nt = 0; nt < 16; ++nt)
#pragma unroll
      for (int i = 0; i < 8; ++i) oacc[nt][i] *= alpha[i];

    // ---- P: C-layout -> LDS -> A-layout fragment (wave-private) ----
#pragma unroll
    for (int i = 0; i < 8; ++i) {
      pb[(i + 8 * lh) * PPAD + ln]      = (bf16)sa[i];
      pb[(i + 8 * lh) * PPAD + 16 + ln] = (bf16)sb[i];
    }
    __builtin_amdgcn_wave_barrier();
    AFrag pf;
    pf.h[0] = *(const bf16x8*)(pb + ln * PPAD + lh * 16);
    pf.h[1] = *(const bf16x8*)(pb + ln * PPAD + lh * 16 + 8);
    __builtin_amdgcn_wave_barrier();

    // ---- O += P @ V over 16 column tiles of F ----
#pragma unroll
    for (int nt = 0; nt < 16; ++nt) {
      bf16x16 vf = *(const bf16x16*)&sV[st][(nt * 16 + ln) * VPAD + lh * 16];
      oacc[nt] = wmma_bf16(pf.v, vf, oacc[nt]);
    }
  }

  // ---- epilogue: O /= l, store bf16 to Ab [B,T,HF] ----
  const int b = bh >> 3, h = bh & 7;
#pragma unroll
  for (int i = 0; i < 8; ++i) {
    const float inv = 1.0f / lrow[i];
    const int tg = qt * 16 + i + 8 * lh;
    const size_t rowoff = ((size_t)(b * T + tg)) * HF + h * F;
#pragma unroll
    for (int nt = 0; nt < 16; ++nt)
      Ab[rowoff + nt * 16 + ln] = (bf16)(oacc[nt][i] * inv);
  }
}

// ---------------------------------------------------------------------------
// Kernel C: projection  out[8192,256] = Ab[8192,2048] @ Wproj + bias (fp32 out)
// ---------------------------------------------------------------------------
__global__ void __launch_bounds__(128) k_proj_gemm(const bf16* __restrict__ Ab,
                                                   const bf16* __restrict__ Wpt,
                                                   const float* __restrict__ bproj,
                                                   float* __restrict__ out) {
  const int wave = threadIdx.x >> 5;
  const int lane = threadIdx.x & 31;
  const int ln = lane & 15, lh = lane >> 4;
  const int m0 = (blockIdx.x * 4 + wave) * 16;
  const int c0 = blockIdx.y * 64;

  f32x8 acc[4];
#pragma unroll
  for (int t = 0; t < 4; ++t)
#pragma unroll
    for (int i = 0; i < 8; ++i) acc[t][i] = 0.0f;

#pragma unroll 4
  for (int ks = 0; ks < HF / 32; ++ks) {
    const int k0 = ks * 32;
    AFrag a;
    const bf16* ap = Ab + (size_t)(m0 + ln) * HF + k0 + lh * 16;
    a.h[0] = *(const bf16x8*)ap;
    a.h[1] = *(const bf16x8*)(ap + 8);
#pragma unroll
    for (int t = 0; t < 4; ++t) {
      bf16x16 b = *(const bf16x16*)(Wpt + (size_t)(c0 + t * 16 + ln) * HF + k0 + lh * 16);
      acc[t] = wmma_bf16(a.v, b, acc[t]);
    }
  }

#pragma unroll
  for (int t = 0; t < 4; ++t) {
    const int c = c0 + t * 16 + ln;
    const float bias = bproj[c];
#pragma unroll
    for (int i = 0; i < 8; ++i) {
      const int m = m0 + i + 8 * lh;
      out[(size_t)m * F + c] = acc[t][i] + bias;
    }
  }
}

// ---------------------------------------------------------------------------
extern "C" void kernel_launch(void* const* d_in, const int* in_sizes, int n_in,
                              void* d_out, int out_size, void* d_ws, size_t ws_size,
                              hipStream_t stream) {
  (void)in_sizes; (void)n_in; (void)out_size; (void)ws_size;
  const float* x     = (const float*)d_in[0];
  const float* Wqkv  = (const float*)d_in[1];
  const float* bqkv  = (const float*)d_in[2];
  const float* Wproj = (const float*)d_in[3];
  const float* bproj = (const float*)d_in[4];
  float* out = (float*)d_out;

  char* ws = (char*)d_ws;
  bf16* xb  = (bf16*)(ws + OFF_XB);
  bf16* Wqt = (bf16*)(ws + OFF_WQT);
  bf16* Wpt = (bf16*)(ws + OFF_WPT);
  bf16* Qb  = (bf16*)(ws + OFF_Q);
  bf16* Kb  = (bf16*)(ws + OFF_K);
  bf16* Vt  = (bf16*)(ws + OFF_VT);
  bf16* Ab  = (bf16*)(ws + OFF_AB);

  {
    const int n = BT * F;
    k_cvt_f32_bf16<<<(n + 255) / 256, 256, 0, stream>>>(x, xb, n);
  }
  k_tr_wqkv<<<HF3, F, 0, stream>>>(Wqkv, Wqt);
  k_tr_wproj<<<F, 256, 0, stream>>>(Wproj, Wpt);

  k_qkv_gemm<<<dim3(BT / 64, HF3 / 64), 128, 0, stream>>>(xb, Wqt, bqkv, Qb, Kb, Vt);

  k_attn<<<(Bsz * H * (T / 16)) / 4, 128, 0, stream>>>(Qb, Kb, Vt, Ab);

  k_proj_gemm<<<dim3(BT / 64, F / 64), 128, 0, stream>>>(Ab, Wpt, bproj, out);
}
